// GeometricLinearizedAttention_24653112279107
// MI455X (gfx1250) — matrix operand order repaired and checked by reading, not compiled, not run on hardware
//
#include <hip/hip_runtime.h>
#include <hip/hip_bf16.h>
#include <math.h>

typedef __attribute__((ext_vector_type(16))) __bf16 v16bf;
typedef __attribute__((ext_vector_type(4)))  __bf16 v4bf;
typedef __attribute__((ext_vector_type(8)))  float  v8f;

#define BH_TOT   64
#define NROWS    8192
#define DH       64
#define SLABS    8
#define RPB      (NROWS / SLABS)   // 1024 rows per block
#define CHUNK    64                // rows per pass1 iteration (2 WMMA k-steps)
#define EPSV     1e-5f
#define CLAMP_E  1e-10f
#define PROJ_E   0.004f

__device__ __forceinline__ float frcp(float x) { return __builtin_amdgcn_rcpf(x); }
__device__ __forceinline__ float eluf1(float x) {   // elu(x)+1
  return (x > 0.f) ? (x + 1.f) : __expf(x);
}

// A-fragment inverse map: element (kk in 0..31, col) of a 16x32 bf16 A-tile
// lives at lane = (col&15) + 16*((kk>>3)&1), slot = (kk&7) | ((kk>>4)<<3)
__device__ __forceinline__ int a_hl(int kk)   { return (kk >> 3) & 1; }
__device__ __forceinline__ int a_slot(int kk) { return (kk & 7) | ((kk & 16) >> 1); }
// B-fragment inverse map: element (kk in 0..31, col) of a 32x16 bf16 B-tile
// lives at lane = (col&15) + 16*((kk>>3)&1), slot = (kk&7) | ((kk>>4)<<3)
__device__ __forceinline__ int b_hl(int kk)   { return (kk >> 3) & 1; }
__device__ __forceinline__ int b_slot(int kk) { return (kk & 7) | ((kk & 16) >> 1); }

// ---------------- K0: zero context + v2_sum ----------------
__global__ void gla_zero(float* p, int n) {
  int i = blockIdx.x * blockDim.x + threadIdx.x;
  if (i < n) p[i] = 0.f;
}

// ---------------- K1: context = v2^T @ x, v2_sum, d0 ----------------
__global__ __launch_bounds__(512)
void gla_pass1(const float* __restrict__ Kp, const float* __restrict__ Vp,
               const float* __restrict__ maskp, const float* __restrict__ curv,
               float* __restrict__ ctx, float* __restrict__ v2sum,
               float* __restrict__ d0out) {
  const int bh   = blockIdx.y;
  const float k  = curv[bh & 15];
  const int t    = threadIdx.x;
  const int lane = t & 31;
  const int wave = t >> 5;
  const int dt   = wave >> 2;       // d-tile 0..3
  const int et   = wave & 3;        // e-tile 0..3

  // double-buffered LDS tiles in WMMA fragment layout: [buf][kc][tile][lane][slot]
  __shared__ __align__(32) __bf16 aV2[2][2][4][32][16];   // 16 KB
  __shared__ __align__(32) __bf16 bX [2][2][4][32][16];   // 16 KB
  __shared__ float sSum[DH];
  if (t < DH) sSum[t] = 0.f;

  // compute-phase mapping: 8 threads per row, 8 cols each
  const int cr = t >> 3;            // row in chunk 0..63
  const int cc = (t & 7) * 8;       // col base
  const int kc  = cr >> 5;          // which 32-row K-step
  const int kkk = cr & 31;
  const int hlA = a_hl(kkk),  ai = a_slot(kkk);
  const int hlB = b_hl(kkk),  bi = b_slot(kkk);

  const size_t base = (size_t)bh * NROWS * DH;
  const float* Kb = Kp + base;
  const float* Vb = Vp + base;

  float v2p[8] = {0.f,0.f,0.f,0.f,0.f,0.f,0.f,0.f};
  v8f acc = {0.f,0.f,0.f,0.f,0.f,0.f,0.f,0.f};

  const int rowBase = blockIdx.x * RPB;
  int ib = 0;

  for (int c0 = 0; c0 < RPB; c0 += CHUNK, ib ^= 1) {
    const int row = rowBase + c0 + cr;
    const float4 va = *(const float4*)(Vb + (size_t)row * DH + cc);
    const float4 vb = *(const float4*)(Vb + (size_t)row * DH + cc + 4);
    const float4 ka = *(const float4*)(Kb + (size_t)row * DH + cc);
    const float4 kb = *(const float4*)(Kb + (size_t)row * DH + cc + 4);
    // prefetch next chunk (speculative; OOB prefetches dropped by HW)
    __builtin_prefetch(Vb + (size_t)(row + CHUNK) * DH + cc, 0, 3);
    __builtin_prefetch(Kb + (size_t)(row + CHUNK) * DH + cc, 0, 3);

    float ss = va.x*va.x + va.y*va.y + va.z*va.z + va.w*va.w
             + vb.x*vb.x + vb.y*vb.y + vb.z*vb.z + vb.w*vb.w;
    ss += __shfl_xor(ss, 1, 32);
    ss += __shfl_xor(ss, 2, 32);
    ss += __shfl_xor(ss, 4, 32);

    const float d0    = fmaxf(1.f + k * ss, 1e-15f);
    const float invd0 = frcp(d0);
    const float gamma = 2.f * invd0;
    const float gm1   = gamma - 1.f;
    const float denom = ((gm1 >= 0.f) ? 1.f : -1.f) * fmaxf(fabsf(gm1), CLAMP_E);
    const float m     = maskp[row];
    const float gd    = gamma * frcp(denom) * m;
    const float dm    = denom * m;
    if ((t & 7) == 0) d0out[(size_t)bh * NROWS + row] = d0;

    const float kv[8] = {ka.x, ka.y, ka.z, ka.w, kb.x, kb.y, kb.z, kb.w};
    const float vv[8] = {va.x, va.y, va.z, va.w, vb.x, vb.y, vb.z, vb.w};
#pragma unroll
    for (int i = 0; i < 8; ++i) {
      const int col = cc + i;
      const float v2e = dm * eluf1(kv[i] * invd0);
      const float xe  = gd * vv[i];
      aV2[ib][kc][col >> 4][(col & 15) + 16 * hlA][ai] = (__bf16)v2e;
      bX [ib][kc][col >> 4][(col & 15) + 16 * hlB][bi] = (__bf16)xe;
      v2p[i] += v2e;
    }
    __syncthreads();   // single barrier: double buffering covers WAR on LDS

#pragma unroll
    for (int kq = 0; kq < 2; ++kq) {
      const v16bf a = *(const v16bf*)aV2[ib][kq][dt][lane];
      const v16bf b = *(const v16bf*)bX [ib][kq][et][lane];
      acc = __builtin_amdgcn_wmma_f32_16x16x32_bf16(false, a, false, b,
                                                    (short)0, acc, false, false);
    }
  }

  // v2_sum: LDS reduce then global atomics
  __syncthreads();
#pragma unroll
  for (int i = 0; i < 8; ++i) atomicAdd(&sSum[cc + i], v2p[i]);
  __syncthreads();
  if (t < DH) atomicAdd(&v2sum[bh * DH + t], sSum[t]);

  // context tile: C/D layout -> VGPR r: lanes0-15 M=r, lanes16-31 M=r+8
  const int hl = lane >> 4, c16 = lane & 15;
#pragma unroll
  for (int r = 0; r < 8; ++r) {
    const int M = r + hl * 8;
    atomicAdd(&ctx[(size_t)bh * 4096 + (dt * 16 + M) * 64 + et * 16 + c16],
              acc[r]);
  }
}

// ---------------- K2: X = (v1*Dinv) @ context, project, mobius, project ----
__global__ __launch_bounds__(512)
void gla_pass2(const float* __restrict__ Qp, const float* __restrict__ curv,
               const float* __restrict__ ctx, const float* __restrict__ v2sum,
               const float* __restrict__ d0buf, float* __restrict__ outp) {
  const int bh   = blockIdx.y;
  const float k  = curv[bh & 15];
  const int t    = threadIdx.x;
  const int lane = t & 31;
  const int wave = t >> 5;
  const int hl   = lane >> 4;
  const int c16  = lane & 15;

  __shared__ __align__(32) __bf16 sctxw[2][4][32][16];     // 8 KB (frag layout)
  // per-wave 4 KB scratch: overlays bf16 A-fragments (2 KB) and f32 C-tile (4 KB)
  __shared__ __align__(32) unsigned char swb[16][4096];    // 64 KB
  __bf16* sv1 = (__bf16*)swb[wave];   // [kc][32][16] bf16 fragment layout
  float*  sxf = (float*)swb[wave];    // [16][64] f32 row-major C tile

  // load context, pre-swizzled into B-fragment layout
  for (int idx = t; idx < 4096; idx += 512) {
    const int d = idx >> 6, n = idx & 63;
    const int kq = d >> 5, kk = d & 31;
    sctxw[kq][n >> 4][(n & 15) + 16 * b_hl(kk)][b_slot(kk)] =
        (__bf16)ctx[(size_t)bh * 4096 + idx];
  }
  __syncthreads();   // sctxw read-only from here on; loop below is barrier-free

  // per-lane v2_sum slice (32 values) held in registers
  const int cbase = hl * 32;
  float4 v2r[8];
#pragma unroll
  for (int j = 0; j < 8; ++j)
    v2r[j] = *(const float4*)(v2sum + bh * DH + cbase + j * 4);

  const float* Qb = Qp + (size_t)bh * NROWS * DH;
  const float sk      = sqrtf(fabsf(k) + 1e-15f);
  const float maxnorm = (k < 0.f) ? (1.f - PROJ_E) / sk : 1e15f;
  const int rowBase   = blockIdx.x * RPB;

  for (int it = 0; it < RPB; it += 256) {
    const int r0  = rowBase + it + wave * 16;
    const int row = r0 + c16;
    const float invd0 = frcp(d0buf[(size_t)bh * NROWS + row]);

    // v1 for this lane's 32 columns + Dn dot product
    float4 e4[8];
    float dnp = 0.f;
#pragma unroll
    for (int j = 0; j < 8; ++j) {
      const float4 q4 = *(const float4*)(Qb + (size_t)row * DH + cbase + j * 4);
      e4[j].x = eluf1(q4.x * invd0);
      e4[j].y = eluf1(q4.y * invd0);
      e4[j].z = eluf1(q4.z * invd0);
      e4[j].w = eluf1(q4.w * invd0);
      dnp += e4[j].x * v2r[j].x + e4[j].y * v2r[j].y
           + e4[j].z * v2r[j].z + e4[j].w * v2r[j].w;
    }
    __builtin_prefetch(Qb + (size_t)(row + 256) * DH + cbase, 0, 3);
    dnp += __shfl_xor(dnp, 16, 32);
    const float dinv = frcp((dnp == 0.f) ? EPSV : dnp);

    // store A fragment = v1 * Dinv (folds the per-row scale into the GEMM)
#pragma unroll
    for (int j = 0; j < 8; ++j) {
      const int kkk = j * 4;
      v4bf p;
      p[0] = (__bf16)(e4[j].x * dinv);
      p[1] = (__bf16)(e4[j].y * dinv);
      p[2] = (__bf16)(e4[j].z * dinv);
      p[3] = (__bf16)(e4[j].w * dinv);
      *(v4bf*)&sv1[((hl * 32) + (c16 + 16 * a_hl(kkk))) * 16 + a_slot(kkk)] = p;
    }

    const v16bf a0 = *(const v16bf*)&sv1[(0 * 32 + lane) * 16];
    const v16bf a1 = *(const v16bf*)&sv1[(1 * 32 + lane) * 16];
#pragma unroll
    for (int e = 0; e < 4; ++e) {
      v8f av = {0.f,0.f,0.f,0.f,0.f,0.f,0.f,0.f};
      const v16bf b0 = *(const v16bf*)sctxw[0][e][lane];
      const v16bf b1 = *(const v16bf*)sctxw[1][e][lane];
      av = __builtin_amdgcn_wmma_f32_16x16x32_bf16(false, a0, false, b0,
                                                   (short)0, av, false, false);
      av = __builtin_amdgcn_wmma_f32_16x16x32_bf16(false, a1, false, b1,
                                                   (short)0, av, false, false);
      // transpose C fragment to row-major f32 in wave-private LDS
#pragma unroll
      for (int r = 0; r < 8; ++r)
        sxf[(r + 8 * hl) * DH + e * 16 + c16] = av[r];
    }

    // tail: lane owns row c16, columns cbase..cbase+31 (in-lane, no butterflies)
    float4 xr[8];
    float s0 = 0.f;
#pragma unroll
    for (int j = 0; j < 8; ++j) {
      xr[j] = *(const float4*)&sxf[c16 * DH + cbase + j * 4];
      s0 += xr[j].x * xr[j].x + xr[j].y * xr[j].y
          + xr[j].z * xr[j].z + xr[j].w * xr[j].w;
    }
    s0 += __shfl_xor(s0, 16, 32);

    const float norm = sqrtf(s0);
    // project #1
    const float s1 = (norm > maxnorm) ? maxnorm * frcp(fmaxf(norm, 1e-15f)) : 1.f;
    const float n2 = norm * s1;
    // mobius scalar mul 0.5 (k<0 here -> atanh/tanh path)
    const float xn = fmaxf(n2, 1e-15f);
    const float su = sk * xn;
    float ar;
    if (k < 0.f) {
      const float sc = fminf(fmaxf(su, -1.f + 1e-7f), 1.f - 1e-7f);
      ar = atanhf(sc);
    } else {
      ar = atanf(su);
    }
    const float st  = 0.5f * ar;                      // sk*(0.5*ar/sk)
    const float tk  = ((k < 0.f) ? tanhf(st) : tanf(st)) * frcp(sk);
    const float fac = tk * frcp(xn);
    const float n3  = n2 * fac;
    // project #2
    const float s2 = (n3 > maxnorm) ? maxnorm * frcp(fmaxf(n3, 1e-15f)) : 1.f;
    const float tot = s1 * fac * s2;

    float* op = outp + ((size_t)bh * NROWS + row) * DH + cbase;
#pragma unroll
    for (int j = 0; j < 8; ++j) {
      float4 o;
      o.x = xr[j].x * tot; o.y = xr[j].y * tot;
      o.z = xr[j].z * tot; o.w = xr[j].w * tot;
      *(float4*)(op + j * 4) = o;
    }
  }
}

extern "C" void kernel_launch(void* const* d_in, const int* in_sizes, int n_in,
                              void* d_out, int out_size, void* d_ws, size_t ws_size,
                              hipStream_t stream) {
  const float* Q    = (const float*)d_in[0];
  const float* K    = (const float*)d_in[1];
  const float* V    = (const float*)d_in[2];
  const float* mask = (const float*)d_in[3];
  const float* curv = (const float*)d_in[4];
  float* out = (float*)d_out;

  float* ws    = (float*)d_ws;
  float* ctx   = ws;                       // 64*64*64      = 262144 floats
  float* v2sum = ws + 262144;              // 64*64         =   4096 floats
  float* d0buf = ws + 266240;              // 64*8192       = 524288 floats

  const int nz = 262144 + 4096;            // must re-zero every call
  gla_zero<<<(nz + 511) / 512, 512, 0, stream>>>(ws, nz);

  dim3 grid(SLABS, BH_TOT);
  gla_pass1<<<grid, 512, 0, stream>>>(K, V, mask, curv, ctx, v2sum, d0buf);
  gla_pass2<<<grid, 512, 0, stream>>>(Q, curv, ctx, v2sum, d0buf, out);
}
